// Attention_47811575939170
// MI455X (gfx1250) — compile-verified, matrix operand order checked by reading, not run on hardware
//
#include <hip/hip_runtime.h>
#include <hip/hip_bf16.h>

// ---------------------------------------------------------------------------
// ViT attention for MI455X (gfx1250): bf16 WMMA 16x16x32, fp32 accumulate.
//   B=64, N=197, D=768, heads=12, dh=64, inner=768, 3*inner=2304
//   tokens T = B*N = 12608 (exactly 788 tiles of 16)
//   attention j padded to 224 (7 tiles of 32)
// Pipeline:
//   0) fp32 -> bf16 conversion pass for x / w_qkv / w_out (done ONCE)
//   1) QKV GEMM (wave tile 32x64), scatter to head-major Q (pre-scaled), K, V^T
//   2) per-(b,h,itile) attention wave: QK^T wmma -> LDS softmax -> P.V^T wmma
//      (V^T tile staged to LDS with async-copy when the builtin exists)
//   3) out GEMM + bias -> fp32
// ---------------------------------------------------------------------------

typedef __attribute__((ext_vector_type(16))) __bf16 v16bf;
typedef __attribute__((ext_vector_type(8)))  __bf16 v8bf;
typedef __attribute__((ext_vector_type(8)))  float  v8f;
typedef __attribute__((ext_vector_type(4)))  int    v4i;

#define HEADS    12
#define DIMH     64
#define SEQ      197
#define SEQP     224
#define BATCH    64
#define DMODEL   768
#define TOKENS   (BATCH * SEQ)   // 12608

#if defined(__AMDGCN__) && \
    __has_builtin(__builtin_amdgcn_global_load_async_to_lds_b128) && \
    __has_builtin(__builtin_amdgcn_s_wait_asynccnt)
#define USE_ASYNC_LDS 1
typedef __attribute__((address_space(1))) v4i* gptr_v4i;
typedef __attribute__((address_space(3))) v4i* lptr_v4i;
#else
#define USE_ASYNC_LDS 0
#endif

__device__ __forceinline__ v8f wmma_bf16(v16bf a, v16bf b, v8f c) {
  return __builtin_amdgcn_wmma_f32_16x16x32_bf16(
      /*neg_a=*/false, a, /*neg_b=*/false, b,
      /*c_mod=*/(short)0, c, /*reuse_a=*/false, /*reuse_b=*/false);
}

// 16x32 WMMA operand fragment (A, or B from an [N x K] row-major source),
// CDNA5 16-bit operand layout (CDNA5 ISA 7.12.2):
//   lanes 0-15 : row = lane,    elems 0..7 = K 0..7,  8..15 = K 16..23
//   lanes 16-31: row = lane-16, elems 0..7 = K 8..15, 8..15 = K 24..31
// Two aligned 16-byte vector loads -> guaranteed b128 (global or ds).
__device__ __forceinline__ v16bf load_frag_bf(const __bf16* base, int ld) {
  const int lane = threadIdx.x & 31;
  const int row  = lane & 15;
  const int kb   = (lane >> 4) << 3;   // 0 or 8
  const __bf16* p = base + (size_t)row * ld + kb;
  v8bf lo = *(const v8bf*)(p);
  v8bf hi = *(const v8bf*)(p + 16);
  return __builtin_shufflevector(lo, hi, 0, 1, 2, 3, 4, 5, 6, 7,
                                 8, 9, 10, 11, 12, 13, 14, 15);
}

// ---------------------------------------------------------------------------
// Kernel 0: fp32 -> bf16, 8 elements per thread.
// ---------------------------------------------------------------------------
__global__ void cvt_bf16_kernel(const float* __restrict__ src,
                                __bf16* __restrict__ dst, int n8) {
  const int i = blockIdx.x * blockDim.x + threadIdx.x;
  if (i < n8) {
    const float4* s = (const float4*)src + (size_t)2 * i;
    float4 f0 = s[0], f1 = s[1];
    v8bf o;
    o[0] = (__bf16)f0.x; o[1] = (__bf16)f0.y;
    o[2] = (__bf16)f0.z; o[3] = (__bf16)f0.w;
    o[4] = (__bf16)f1.x; o[5] = (__bf16)f1.y;
    o[6] = (__bf16)f1.z; o[7] = (__bf16)f1.w;
    *((v8bf*)dst + i) = o;
  }
}

// ---------------------------------------------------------------------------
// Kernel 1: QKV projection. [12608 x 768] x [768 x 2304] (w stored [out,in]).
// 4 waves/block, wave tile 32x64 -> block tile 64x128. 768/128 = 6 blocks per
// Q/K/V region, so `which` is block-uniform (scalar branch in the scatter).
// ---------------------------------------------------------------------------
__global__ void qkv_kernel(const __bf16* __restrict__ xb,
                           const __bf16* __restrict__ wb,
                           __bf16* __restrict__ Q,
                           __bf16* __restrict__ K,
                           __bf16* __restrict__ Vt) {
  const int lane = threadIdx.x & 31;
  const int wave = threadIdx.x >> 5;            // 0..3
  const int wm = wave >> 1, wn = wave & 1;
  const int mbase = blockIdx.y * 64 + wm * 32;  // token row
  const int col0  = blockIdx.x * 128 + wn * 64; // col in [0,2304)

  v8f acc[2][4] = {};
  for (int k = 0; k < DMODEL; k += 32) {
    const __bf16* pa = xb + (size_t)mbase * DMODEL + k;
    const __bf16* pb = wb + (size_t)col0 * DMODEL + k;
    __builtin_prefetch(pa + 32, 0, 0);
    __builtin_prefetch(pb + 32, 0, 0);
    v16bf a0 = load_frag_bf(pa, DMODEL);
    v16bf a1 = load_frag_bf(pa + 16 * DMODEL, DMODEL);
#pragma unroll
    for (int ni = 0; ni < 4; ++ni) {
      v16bf b = load_frag_bf(pb + (size_t)ni * 16 * DMODEL, DMODEL);
      acc[0][ni] = wmma_bf16(a0, b, acc[0][ni]);
      acc[1][ni] = wmma_bf16(a1, b, acc[1][ni]);
    }
  }

  const int which  = blockIdx.x / 6;                  // uniform: 0=Q 1=K 2=V
  const int iobase = col0 - which * DMODEL;           // col within [0,768)
  const int hi = lane >> 4, nn = lane & 15;
#pragma unroll
  for (int mi = 0; mi < 2; ++mi)
#pragma unroll
    for (int ni = 0; ni < 4; ++ni)
#pragma unroll
      for (int r = 0; r < 8; ++r) {
        const int t  = mbase + mi * 16 + r + 8 * hi;  // token, < 12608
        const int io = iobase + ni * 16 + nn;         // inner col, < 768
        const float v = acc[mi][ni][r];
        const int b = t / SEQ, n = t - b * SEQ;
        const int h = io >> 6, d = io & 63;
        const size_t bh = (size_t)(b * HEADS + h);
        if (which == 0)
          Q[(bh * SEQP + n) * DIMH + d] = (__bf16)(v * 0.125f);  // 1/sqrt(64)
        else if (which == 1)
          K[(bh * SEQP + n) * DIMH + d] = (__bf16)v;
        else
          Vt[(bh * DIMH + d) * SEQP + n] = (__bf16)v;            // transposed
      }
}

// ---------------------------------------------------------------------------
// Kernel 2: attention, one wave per (b, h, 16-row Q tile).
// ---------------------------------------------------------------------------
__global__ void attn_kernel(const __bf16* __restrict__ Q,
                            const __bf16* __restrict__ K,
                            const __bf16* __restrict__ Vt,
                            __bf16* __restrict__ ctx) {
  __shared__ float  sS[16][SEQP];      // 14336 B
  __shared__ __bf16 sP[16][SEQP];      //  7168 B
  __shared__ float  sSum[16];
#if USE_ASYNC_LDS
  __shared__ __bf16 sV[DIMH * SEQP];   // 28672 B
#endif

  const int bid = blockIdx.x;
  const int it  = bid % 13;            // 13 i-tiles cover 197 rows
  const int bh  = bid / 13;            // b*HEADS + h
  const int lane = threadIdx.x;        // block = 32 = one wave
  const int hi = lane >> 4, nn = lane & 15;

  const __bf16* Qb = Q  + ((size_t)bh * SEQP + it * 16) * DIMH;
  const __bf16* Kb = K  + (size_t)bh * SEQP * DIMH;
  const __bf16* Vb = Vt + (size_t)bh * DIMH * SEQP;

#if USE_ASYNC_LDS
  // Kick off the async V^T tile copy now; it overlaps QK^T + softmax and is
  // waited on (ASYNCcnt) right before the P.V^T phase.
#pragma unroll 4
  for (int ofs = lane * 8; ofs < DIMH * SEQP; ofs += 32 * 8) {
    __builtin_amdgcn_global_load_async_to_lds_b128(
        (gptr_v4i)(v4i*)(Vb + ofs),
        (lptr_v4i)(v4i*)(sV + ofs), 0, 0);
  }
#endif

  // Q tile fragments: 16 x 64 = two 16x32 A fragments
  v16bf qa0 = load_frag_bf(Qb + 0,  DIMH);
  v16bf qa1 = load_frag_bf(Qb + 32, DIMH);

  // ---- scores: 14 j-tiles of 16 columns (224 total) ----
  for (int jt = 0; jt < 14; ++jt) {
    v8f s = {};
    s = wmma_bf16(qa0, load_frag_bf(Kb + (size_t)jt * 16 * DIMH + 0,  DIMH), s);
    s = wmma_bf16(qa1, load_frag_bf(Kb + (size_t)jt * 16 * DIMH + 32, DIMH), s);
    const int j = jt * 16 + nn;
#pragma unroll
    for (int r = 0; r < 8; ++r)
      sS[r + 8 * hi][j] = (j < SEQ) ? s[r] : -__builtin_inff();
  }
  __syncthreads();

  // ---- softmax: 2 lanes per row ----
  {
    const int row = lane >> 1, half = lane & 1;
    float m = -__builtin_inff();
    for (int j = half; j < SEQP; j += 2) m = fmaxf(m, sS[row][j]);
    m = fmaxf(m, __shfl_xor(m, 1, 32));
    float sum = 0.0f;
    for (int j = half; j < SEQP; j += 2) {
      float p = __expf(sS[row][j] - m);   // exp(-inf)=0 for masked j
      sum += p;
      sP[row][j] = (__bf16)p;             // un-normalized probs
    }
    sum += __shfl_xor(sum, 1, 32);
    if (half == 0) sSum[row] = sum;
  }
  __syncthreads();

#if USE_ASYNC_LDS
  __builtin_amdgcn_s_wait_asynccnt(0);
  const __bf16* Vsrc = sV;
#else
  const __bf16* Vsrc = Vb;
#endif

  // ---- out = P . V^T : K-dim = 224 in 7 steps of 32, d in 4 tiles of 16 ----
  v8f o[4] = {};
  for (int jt = 0; jt < 7; ++jt) {
    v16bf pa = load_frag_bf(&sP[0][0] + jt * 32, SEQP);
#pragma unroll
    for (int dt = 0; dt < 4; ++dt) {
      v16bf vb = load_frag_bf(Vsrc + (size_t)dt * 16 * SEQP + jt * 32, SEQP);
      o[dt] = wmma_bf16(pa, vb, o[dt]);
    }
  }

  const int b = bh / HEADS, h = bh - (bh / HEADS) * HEADS;
#pragma unroll
  for (int dt = 0; dt < 4; ++dt)
#pragma unroll
    for (int r = 0; r < 8; ++r) {
      const int row = r + 8 * hi;
      const int n = it * 16 + row;
      if (n < SEQ) {
        const float inv = 1.0f / sSum[row];
        ctx[((size_t)(b * SEQ + n)) * DMODEL + h * DIMH + dt * 16 + nn] =
            (__bf16)(o[dt][r] * inv);
      }
    }
}

// ---------------------------------------------------------------------------
// Kernel 3: out projection. [12608 x 768] x [768 x 768] + bias -> fp32.
// Same 32x64 wave tiles as kernel 1.
// ---------------------------------------------------------------------------
__global__ void out_kernel(const __bf16* __restrict__ ctx,
                           const __bf16* __restrict__ wb,
                           const float* __restrict__ bias,
                           float* __restrict__ out) {
  const int lane = threadIdx.x & 31;
  const int wave = threadIdx.x >> 5;
  const int wm = wave >> 1, wn = wave & 1;
  const int mbase = blockIdx.y * 64 + wm * 32;
  const int col0  = blockIdx.x * 128 + wn * 64;

  v8f acc[2][4] = {};
  for (int k = 0; k < DMODEL; k += 32) {
    const __bf16* pa = ctx + (size_t)mbase * DMODEL + k;
    const __bf16* pb = wb  + (size_t)col0 * DMODEL + k;
    __builtin_prefetch(pa + 32, 0, 0);
    __builtin_prefetch(pb + 32, 0, 0);
    v16bf a0 = load_frag_bf(pa, DMODEL);
    v16bf a1 = load_frag_bf(pa + 16 * DMODEL, DMODEL);
#pragma unroll
    for (int ni = 0; ni < 4; ++ni) {
      v16bf b = load_frag_bf(pb + (size_t)ni * 16 * DMODEL, DMODEL);
      acc[0][ni] = wmma_bf16(a0, b, acc[0][ni]);
      acc[1][ni] = wmma_bf16(a1, b, acc[1][ni]);
    }
  }

  const int hi = lane >> 4, nn = lane & 15;
#pragma unroll
  for (int mi = 0; mi < 2; ++mi)
#pragma unroll
    for (int ni = 0; ni < 4; ++ni)
#pragma unroll
      for (int r = 0; r < 8; ++r) {
        const int t = mbase + mi * 16 + r + 8 * hi;
        const int o = col0 + ni * 16 + nn;
        out[(size_t)t * DMODEL + o] = acc[mi][ni][r] + bias[o];
      }
}

// ---------------------------------------------------------------------------
// Workspace layout (bytes), all offsets multiples of 256:
//   xb    [12608 x 768] bf16      @ 0            (19,365,888)
//   wqkvb [2304 x 768]  bf16      @ 19,365,888   ( 3,538,944)
//   woutb [768 x 768]   bf16      @ 22,904,832   ( 1,179,648)
//   Q     [768][224][64] bf16     @ 24,084,480   (22,020,096)
//   K     same                    @ 46,104,576
//   Vt    [768][64][224] bf16     @ 68,124,672
//   ctx   [12608 x 768] bf16      @ 90,144,768   (19,365,888)
// Total ~109.5 MB.
// ---------------------------------------------------------------------------
extern "C" void kernel_launch(void* const* d_in, const int* in_sizes, int n_in,
                              void* d_out, int out_size, void* d_ws, size_t ws_size,
                              hipStream_t stream) {
  const float* x     = (const float*)d_in[0];
  const float* w_qkv = (const float*)d_in[1];
  const float* w_out = (const float*)d_in[2];
  const float* b_out = (const float*)d_in[3];
  float* out = (float*)d_out;

  char* ws = (char*)d_ws;
  __bf16* xb    = (__bf16*)(ws + 0);
  __bf16* wqkvb = (__bf16*)(ws + 19365888);
  __bf16* woutb = (__bf16*)(ws + 22904832);
  __bf16* Q     = (__bf16*)(ws + 24084480);
  __bf16* K     = (__bf16*)(ws + 46104576);
  __bf16* Vt    = (__bf16*)(ws + 68124672);
  __bf16* ctx   = (__bf16*)(ws + 90144768);

  // 0) one-time fp32 -> bf16 conversion of activations and weights
  {
    int n8;
    n8 = TOKENS * DMODEL / 8;           // 1,210,368
    cvt_bf16_kernel<<<(n8 + 255) / 256, 256, 0, stream>>>(x, xb, n8);
    n8 = 3 * DMODEL * DMODEL / 8;       //   221,184
    cvt_bf16_kernel<<<(n8 + 255) / 256, 256, 0, stream>>>(w_qkv, wqkvb, n8);
    n8 = DMODEL * DMODEL / 8;           //    73,728
    cvt_bf16_kernel<<<(n8 + 255) / 256, 256, 0, stream>>>(w_out, woutb, n8);
  }

  // 1) QKV projection: 12608 x 2304 in 64x128 block tiles
  qkv_kernel<<<dim3(2304 / 128, TOKENS / 64), 128, 0, stream>>>(xb, wqkvb, Q, K, Vt);

  // 2) attention: one wave per (b, h, i-tile); 64*12*13 = 9984 blocks
  attn_kernel<<<dim3(BATCH * HEADS * 13), 32, 0, stream>>>(Q, K, Vt, ctx);

  // 3) output projection: 12608 x 768 in 64x128 block tiles
  out_kernel<<<dim3(DMODEL / 128, TOKENS / 64), 128, 0, stream>>>(ctx, woutb, b_out, out);
}
